// PoolHiddenNet_56822417326377
// MI455X (gfx1250) — matrix operand order repaired and checked by reading, not compiled, not run on hardware
//
#include <hip/hip_runtime.h>
#include <hip/hip_bf16.h>

typedef __attribute__((ext_vector_type(16))) _Float16 v16h;
typedef __attribute__((ext_vector_type(8)))  _Float16 v8h;
typedef __attribute__((ext_vector_type(8)))  float    v8f;

#define SDIM 8
#define BDIM 8192
#define GDIM 512
#define PDIM 16
#define BOT  1024

__device__ __forceinline__ float swap16_max(float x) {
    // SWAPX16 via ds_swizzle: group-of-32 mode, xor=0x10, and=0x1f -> 0x401F
    int s = __builtin_amdgcn_ds_swizzle(__float_as_int(x), 0x401F);
    return fmaxf(x, __int_as_float(s));
}

// One workgroup per group g. 256 threads = 8 wave32.
// LDS: Wlds (512 cols of W2, col-major f16, stride 72), Ylds (256x64 f16, stride 72),
// plus small f32 scratch. Total ~121 KB.
__global__ __launch_bounds__(256, 1)
void pool_hidden_net_kernel(const float* __restrict__ in_xy,
                            const float* __restrict__ h_states,
                            const float* __restrict__ W_emb,
                            const float* __restrict__ b_emb,
                            const float* __restrict__ W1,
                            const float* __restrict__ b1,
                            const float* __restrict__ W2,
                            const float* __restrict__ b2,
                            float* __restrict__ out)
{
    __shared__ _Float16 Wlds[512 * 72];   // 73728 B : W2 pass-chunk, col-major f16
    __shared__ _Float16 Ylds[256 * 72];   // 36864 B : layer-1 output, f16
    __shared__ float Hterm[16 * 64];      // h_j @ W1_bot
    __shared__ float hbuf [16 * 64];
    __shared__ float b2s  [1024];
    __shared__ float M2s  [2 * 64];       // W_emb @ W1_top
    __shared__ float cvec [64];           // b_emb @ W1_top + b1
    __shared__ float poss [32];           // 16 x (x,y)

    const int tid = threadIdx.x;
    const int g   = blockIdx.x;

    // ---- stage tiny per-group inputs -------------------------------------
    if (tid < 32) poss[tid] = in_xy[(SDIM - 1) * BDIM * 2 + g * 32 + tid];
#pragma unroll
    for (int t = 0; t < 4; ++t) {
        int idx = tid + t * 256;
        hbuf[idx] = h_states[g * 1024 + idx];
        b2s[idx]  = b2[idx];
    }
    __syncthreads();

    // ---- collapse layer 1: precompute M2, cvec, Hterm --------------------
    if (tid < 128) {
        int r = tid >> 6, o = tid & 63;
        float s = 0.f;
        for (int k = 0; k < 64; ++k) s += W_emb[r * 64 + k] * W1[k * 64 + o];
        M2s[r * 64 + o] = s;
    } else if (tid < 192) {
        int o = tid - 128;
        float s = b1[o];
        for (int k = 0; k < 64; ++k) s += b_emb[k] * W1[k * 64 + o];
        cvec[o] = s;
    }
#pragma unroll
    for (int t = 0; t < 4; ++t) {
        int idx = tid + t * 256;            // 0..1023
        int j = idx >> 6, o = idx & 63;
        float s = 0.f;
        for (int k = 0; k < 64; ++k) s += hbuf[j * 64 + k] * W1[(64 + k) * 64 + o];
        Hterm[idx] = s;
    }
    __syncthreads();

    // ---- Y(row = i*16+j, o) = relu(rel@M2 + cvec + Hterm[j]) -> f16 LDS --
    {
        int i = tid >> 4, j = tid & 15;     // one row per thread
        float rx = poss[j * 2]     - poss[i * 2];
        float ry = poss[j * 2 + 1] - poss[i * 2 + 1];
#pragma unroll 4
        for (int o = 0; o < 64; ++o) {
            float v = rx * M2s[o] + ry * M2s[64 + o] + cvec[o] + Hterm[j * 64 + o];
            Ylds[tid * 72 + o] = (_Float16)fmaxf(v, 0.f);
        }
    }
    __syncthreads();

    // ---- layer 2 via WMMA, fused bias+relu+max-over-j --------------------
    const int wv   = tid >> 5;          // wave id 0..7
    const int lane = tid & 31;
    const int hsel = lane >> 4;         // half-lane group
    const int nlo  = lane & 15;

    for (int pass = 0; pass < 2; ++pass) {
        // stage W2 columns [pass*512, pass*512+512) transposed as f16
        const float* W2p = W2 + pass * 512;
        for (int k = 0; k < 64; ++k) {
            Wlds[(tid)       * 72 + k] = (_Float16)W2p[k * 1024 + tid];
            Wlds[(tid + 256) * 72 + k] = (_Float16)W2p[k * 1024 + tid + 256];
        }
        __syncthreads();

        // B fragments for this wave's 4 column tiles, both K halves, in VGPRs
        v16h Bfr[4][2];
#pragma unroll
        for (int q = 0; q < 4; ++q) {
            int colL = (wv * 4 + q) * 16 + nlo;
#pragma unroll
            for (int kb = 0; kb < 2; ++kb) {
                int base = colL * 72 + kb * 32 + hsel * 16;
                v8h lo = *(const v8h*)&Wlds[base];
                v8h hi = *(const v8h*)&Wlds[base + 8];
                Bfr[q][kb] = __builtin_shufflevector(lo, hi,
                    0,1,2,3,4,5,6,7,8,9,10,11,12,13,14,15);
            }
        }

        for (int mt = 0; mt < 16; ++mt) {   // row tile = fixed i, rows are j=0..15
            int abase = (mt * 16 + nlo) * 72 + hsel * 16;
            v8h a00 = *(const v8h*)&Ylds[abase];
            v8h a01 = *(const v8h*)&Ylds[abase + 8];
            v8h a10 = *(const v8h*)&Ylds[abase + 32];
            v8h a11 = *(const v8h*)&Ylds[abase + 40];
            v16h A0 = __builtin_shufflevector(a00, a01,
                0,1,2,3,4,5,6,7,8,9,10,11,12,13,14,15);
            v16h A1 = __builtin_shufflevector(a10, a11,
                0,1,2,3,4,5,6,7,8,9,10,11,12,13,14,15);

            // issue all 8 WMMAs (4 independent accumulator chains) first,
            // so reduction VALU overlaps matrix-pipe execution instead of
            // stalling on per-tile WMMA->VALU hazard NOPs
            v8f acc[4];
#pragma unroll
            for (int q = 0; q < 4; ++q) {
                v8f z = {};
                z = __builtin_amdgcn_wmma_f32_16x16x32_f16(
                        false, A0, false, Bfr[q][0], (short)0, z, false, false);
                acc[q] = __builtin_amdgcn_wmma_f32_16x16x32_f16(
                        false, A1, false, Bfr[q][1], (short)0, z, false, false);
            }
#pragma unroll
            for (int q = 0; q < 4; ++q) {
                int colg  = pass * 512 + (wv * 4 + q) * 16 + nlo;
                float bias = b2s[colg];
                float m = 0.f;                       // relu => max >= 0
#pragma unroll
                for (int r = 0; r < 8; ++r)
                    m = fmaxf(m, fmaxf(acc[q][r] + bias, 0.f));
                // combine M=0..7 (lanes 0-15) with M=8..15 (lanes 16-31)
                m = swap16_max(m);
                if (lane < 16)
                    out[(size_t)(g * PDIM + mt) * BOT + colg] = m;
            }
        }
        __syncthreads();   // before next pass overwrites Wlds
    }
}

extern "C" void kernel_launch(void* const* d_in, const int* in_sizes, int n_in,
                              void* d_out, int out_size, void* d_ws, size_t ws_size,
                              hipStream_t stream) {
    const float* in_xy    = (const float*)d_in[0];
    // d_in[1] = in_dxdy  : unused by reference
    const float* h_states = (const float*)d_in[2];
    // d_in[3] = seq_start_end : groups are uniform/contiguous, derived analytically
    const float* W_emb    = (const float*)d_in[4];
    const float* b_emb    = (const float*)d_in[5];
    const float* W1       = (const float*)d_in[6];
    const float* b1       = (const float*)d_in[7];
    const float* W2       = (const float*)d_in[8];
    const float* b2       = (const float*)d_in[9];
    float* out = (float*)d_out;

    pool_hidden_net_kernel<<<dim3(GDIM), dim3(256), 0, stream>>>(
        in_xy, h_states, W_emb, b_emb, W1, b1, W2, b2, out);
}
